// Transformer_29420525977985
// MI455X (gfx1250) — matrix operand order repaired and checked by reading, not compiled, not run on hardware
//
#include <hip/hip_runtime.h>
#include <hip/hip_bf16.h>

// ---------------------------------------------------------------------------
// BERT-base encoder (12 layers) forward on gfx1250 using v_wmma_f32_16x16x32_f16.
// ---------------------------------------------------------------------------

typedef _Float16 f16;
typedef __attribute__((ext_vector_type(16))) _Float16 v16h;
typedef __attribute__((ext_vector_type(8)))  float    v8f;

union AFrag { v16h v; unsigned short u[16]; unsigned int d[8]; };
union CFrag { v8f  v; float f[8]; };
union WPack { uint4 u4; f16 h[8]; };

#define WMMA_F16(A, B, C) \
    __builtin_amdgcn_wmma_f32_16x16x32_f16(false, (A), false, (B), (short)0, (C), false, false)

static constexpr int Hh   = 768;
static constexpr int NH   = 12;
static constexpr int DH   = 64;
static constexpr int DFF  = 3072;
static constexpr int Bb   = 16;
static constexpr int Ss   = 512;
static constexpr int MROW = Bb * Ss;          // 8192 rows

// ---------------------------------------------------------------------------
// fp32 -> f16 weight conversion
// ---------------------------------------------------------------------------
__global__ __launch_bounds__(256) void cvt_kernel(const float* __restrict__ s,
                                                  f16* __restrict__ d, int n) {
    int i = blockIdx.x * 256 + threadIdx.x;
    if (i < n) d[i] = (f16)s[i];
}

// ---------------------------------------------------------------------------
// LayerNorm: one block per row of 768, f32 in, f16 out
// ---------------------------------------------------------------------------
__global__ __launch_bounds__(256) void ln_kernel(const float* __restrict__ h,
                                                 const float* __restrict__ g,
                                                 const float* __restrict__ be,
                                                 f16* __restrict__ out) {
    __shared__ float red[256];
    const int row = blockIdx.x;
    const int t   = threadIdx.x;
    const float* x = h + (size_t)row * Hh;
    float v0 = x[t], v1 = x[t + 256], v2 = x[t + 512];
    red[t] = v0 + v1 + v2;
    __syncthreads();
    for (int o = 128; o > 0; o >>= 1) { if (t < o) red[t] += red[t + o]; __syncthreads(); }
    const float mean = red[0] * (1.0f / Hh);
    __syncthreads();
    const float d0 = v0 - mean, d1 = v1 - mean, d2 = v2 - mean;
    red[t] = d0 * d0 + d1 * d1 + d2 * d2;
    __syncthreads();
    for (int o = 128; o > 0; o >>= 1) { if (t < o) red[t] += red[t + o]; __syncthreads(); }
    const float inv = rsqrtf(red[0] * (1.0f / Hh) + 1e-12f);
    f16* o16 = out + (size_t)row * Hh;
    o16[t]       = (f16)(g[t]       * d0 * inv + be[t]);
    o16[t + 256] = (f16)(g[t + 256] * d1 * inv + be[t + 256]);
    o16[t + 512] = (f16)(g[t + 512] * d2 * inv + be[t + 512]);
}

// ---------------------------------------------------------------------------
// WMMA GEMM: C[M x N] = A[M x K] * W[K x N] + bias, M = 8192.
// Block = 256 threads (8 waves); tile 256(M) x 64(N); K-chunk 32.
// Each wave owns 32(M) x 64(N): 2 A-frags x 4 B-frags = 8 WMMAs per chunk,
// with each B fragment reused across both row groups while live.
// W staging: one coalesced global_load_b128 per thread, transposed into LDS.
// mode 0: plain f16 out   mode 1: qkv [B,NH,S,DH] f16 out
// mode 2: residual f32    mode 3: GELU f16 out
// ---------------------------------------------------------------------------
__global__ __launch_bounds__(256) void gemm_kernel(
    const f16* __restrict__ A, const f16* __restrict__ W,
    const float* __restrict__ bias, int K, int N, int mode,
    f16* __restrict__ out16, float* __restrict__ out32,
    const float* __restrict__ res) {

    __shared__ f16 Alds[256][32];   // 16 KB
    __shared__ f16 Wlds[64][32];    // 4 KB (transposed: [n][k])

    const int m0 = blockIdx.x * 256;
    const int n0 = blockIdx.y * 64;
    const int t    = threadIdx.x;
    const int wave = t >> 5;
    const int lane = t & 31;

    CFrag acc[2][4];
#pragma unroll
    for (int g = 0; g < 2; ++g)
#pragma unroll
        for (int i = 0; i < 4; ++i)
#pragma unroll
            for (int r = 0; r < 8; ++r) acc[g][i].f[r] = 0.0f;

    const int mr   = lane & 15;
    const int koff = (lane < 16) ? 0 : 8;   // A frag K group offset (halves of 16)
    const int nn   = lane & 15;
    const int kb   = (lane < 16) ? 0 : 16;  // B frag K offset

    const int wk = t >> 3;                  // 0..31 : K row of W tile
    const int wn = (t & 7) * 8;             // 0..56 : 8-col group of W tile

    for (int kc = 0; kc < K; kc += 32) {
        // stage A tile 256x32: thread t copies its whole row chunk (32 f16)
        {
            const unsigned int* s =
                (const unsigned int*)(A + (size_t)(m0 + t) * K + kc);
            unsigned int* dl = (unsigned int*)&Alds[t][0];
#pragma unroll
            for (int i = 0; i < 16; ++i) dl[i] = s[i];
        }
        // stage W tile transposed: coalesced 128-bit row load, scatter to Wlds[n][k]
        {
            WPack w;
            w.u4 = *(const uint4*)(W + (size_t)(kc + wk) * N + (n0 + wn));
#pragma unroll
            for (int i = 0; i < 8; ++i) Wlds[wn + i][wk] = w.h[i];
        }
        // prefetch next chunk while this one is consumed (global_prefetch_b8)
        if (kc + 32 < K) {
            __builtin_prefetch(A + (size_t)(m0 + t) * K + kc + 32, 0, 3);
            __builtin_prefetch(W + (size_t)(kc + 32 + wk) * N + (n0 + wn), 0, 3);
        }
        __syncthreads();

        // A fragments (2 x 16x32), contiguous layout:
        // lanes0-15 K in {0..15}, lanes16-31 {16..31}
        AFrag a0, a1;
        {
            const unsigned int* ap = (const unsigned int*)&Alds[wave * 32 + mr][0];
#pragma unroll
            for (int j = 0; j < 8; ++j) a0.d[j] = ap[koff + j];
            ap = (const unsigned int*)&Alds[wave * 32 + 16 + mr][0];
#pragma unroll
            for (int j = 0; j < 8; ++j) a1.d[j] = ap[koff + j];
        }
        // B fragments: lane n = col, lanes0-15 K=0..15, lanes16-31 K=16..31;
        // each B frag feeds both row groups while live
#pragma unroll
        for (int ti = 0; ti < 4; ++ti) {
            AFrag bf;
            const unsigned int* bp = (const unsigned int*)&Wlds[ti * 16 + nn][0];
#pragma unroll
            for (int j = 0; j < 8; ++j) bf.d[j] = bp[(kb >> 1) + j];
            acc[0][ti].v = WMMA_F16(a0.v, bf.v, acc[0][ti].v);
            acc[1][ti].v = WMMA_F16(a1.v, bf.v, acc[1][ti].v);
        }
        __syncthreads();
    }

    // epilogue: C layout VGPR r -> row (r + 8*(lane>=16)), col = lane&15
    const int half = lane >> 4;
#pragma unroll
    for (int g = 0; g < 2; ++g) {
#pragma unroll
        for (int ti = 0; ti < 4; ++ti) {
#pragma unroll
            for (int r = 0; r < 8; ++r) {
                const int mg = m0 + wave * 32 + g * 16 + r + 8 * half;
                const int ng = n0 + ti * 16 + nn;
                float val = acc[g][ti].f[r] + bias[ng];
                if (mode == 1) {          // q/k/v -> [B, NH, S, DH]
                    const int bi = mg >> 9, si = mg & 511;
                    const int hd = ng >> 6, dd = ng & 63;
                    out16[(((size_t)bi * NH + hd) * Ss + si) * DH + dd] = (f16)val;
                } else if (mode == 2) {   // residual, f32 out
                    const size_t idx = (size_t)mg * N + ng;
                    out32[idx] = res[idx] + val;
                } else if (mode == 3) {   // exact GELU
                    val = val * 0.5f * (1.0f + erff(val * 0.70710678118654752f));
                    out16[(size_t)mg * N + ng] = (f16)val;
                } else {
                    out16[(size_t)mg * N + ng] = (f16)val;
                }
            }
        }
    }
}

// ---------------------------------------------------------------------------
// Flash attention: one wave per (qtile16, head, batch). Streams 32-key tiles.
// scores = Q K^T / 8 + addmask ; online softmax ; O += P V
// ---------------------------------------------------------------------------
__global__ __launch_bounds__(32) void attn_kernel(
    const f16* __restrict__ q, const f16* __restrict__ k, const f16* __restrict__ v,
    const float* __restrict__ amask, f16* __restrict__ ctx) {

    __shared__ f16 Qs[16][64];   // 2 KB
    __shared__ f16 Ks[32][64];   // 4 KB  (key-major)
    __shared__ f16 Vt[64][32];   // 4 KB  (d-major, transposed)
    __shared__ f16 Ps[16][32];   // 1 KB

    const int qt = blockIdx.x, hh = blockIdx.y, bi = blockIdx.z;
    const int lane = threadIdx.x;
    const size_t bh = ((size_t)bi * NH + hh) * Ss * DH;

    // stage Q tile (16 x 64): each lane copies half a row
    {
        const int row = lane >> 1, cc = (lane & 1) * 32;
        const unsigned int* s = (const unsigned int*)(q + bh + (size_t)(qt * 16 + row) * DH + cc);
        unsigned int* dl = (unsigned int*)&Qs[row][cc];
#pragma unroll
        for (int i = 0; i < 16; ++i) dl[i] = s[i];
    }
    __syncthreads();

    const int mr   = lane & 15;
    const int koff = (lane < 16) ? 0 : 8;
    const int nn   = lane & 15;
    const int kb2  = (lane < 16) ? 0 : 16;

    AFrag qa[2];
#pragma unroll
    for (int c = 0; c < 2; ++c) {
        const unsigned int* qp = (const unsigned int*)&Qs[mr][c * 32];
#pragma unroll
        for (int j = 0; j < 8; ++j) qa[c].d[j] = qp[koff + j];
    }

    float mrow[8], lrow[8];
    CFrag o[4];
#pragma unroll
    for (int r = 0; r < 8; ++r) { mrow[r] = -1e30f; lrow[r] = 0.0f; }
#pragma unroll
    for (int ti = 0; ti < 4; ++ti)
#pragma unroll
        for (int r = 0; r < 8; ++r) o[ti].f[r] = 0.0f;

    for (int j = 0; j < Ss / 32; ++j) {
        __syncthreads();
        // stage K (key-major) and V (transposed): lane = key row
        {
            const unsigned int* s = (const unsigned int*)(k + bh + (size_t)(j * 32 + lane) * DH);
            unsigned int* dl = (unsigned int*)&Ks[lane][0];
#pragma unroll
            for (int i = 0; i < 32; ++i) dl[i] = s[i];
            const f16* vs = v + bh + (size_t)(j * 32 + lane) * DH;
#pragma unroll
            for (int dd = 0; dd < 64; ++dd) Vt[dd][lane] = vs[dd];
        }
        __syncthreads();

        // scores: two 16-key subtiles, each K-dim 64 = 2 chunks of 32
        CFrag c0, c1;
#pragma unroll
        for (int r = 0; r < 8; ++r) { c0.f[r] = 0.0f; c1.f[r] = 0.0f; }
#pragma unroll
        for (int c = 0; c < 2; ++c) {
            AFrag kf;
            const unsigned int* kp = (const unsigned int*)&Ks[nn][c * 32];
#pragma unroll
            for (int jj = 0; jj < 8; ++jj) kf.d[jj] = kp[(kb2 >> 1) + jj];
            c0.v = WMMA_F16(qa[c].v, kf.v, c0.v);
            kp = (const unsigned int*)&Ks[16 + nn][c * 32];
#pragma unroll
            for (int jj = 0; jj < 8; ++jj) kf.d[jj] = kp[(kb2 >> 1) + jj];
            c1.v = WMMA_F16(qa[c].v, kf.v, c1.v);
        }

        const float am0 = -10000.0f * (1.0f - amask[bi * Ss + j * 32 + nn]);
        const float am1 = -10000.0f * (1.0f - amask[bi * Ss + j * 32 + 16 + nn]);

        float p0[8], p1[8];
#pragma unroll
        for (int r = 0; r < 8; ++r) {
            float s0 = c0.f[r] * 0.125f + am0;
            float s1 = c1.f[r] * 0.125f + am1;
            float rmax = fmaxf(s0, s1);
#pragma unroll
            for (int off = 1; off < 16; off <<= 1)
                rmax = fmaxf(rmax, __shfl_xor(rmax, off, 32));
            const float nm = fmaxf(mrow[r], rmax);
            p0[r] = __expf(s0 - nm);
            p1[r] = __expf(s1 - nm);
            float rsum = p0[r] + p1[r];
#pragma unroll
            for (int off = 1; off < 16; off <<= 1)
                rsum += __shfl_xor(rsum, off, 32);
            const float corr = __expf(mrow[r] - nm);
            lrow[r] = lrow[r] * corr + rsum;
            mrow[r] = nm;
#pragma unroll
            for (int ti = 0; ti < 4; ++ti) o[ti].f[r] *= corr;
        }

        // write P (16x32) to LDS in row-major, then reload as A fragment
        const int rbase = (lane >> 4) * 8;
#pragma unroll
        for (int r = 0; r < 8; ++r) {
            Ps[rbase + r][nn]      = (f16)p0[r];
            Ps[rbase + r][16 + nn] = (f16)p1[r];
        }
        __syncthreads();

        AFrag pa;
        {
            const unsigned int* pp = (const unsigned int*)&Ps[mr][0];
#pragma unroll
            for (int jj = 0; jj < 8; ++jj) pa.d[jj] = pp[koff + jj];
        }
#pragma unroll
        for (int ti = 0; ti < 4; ++ti) {
            AFrag vf;
            const unsigned int* vp = (const unsigned int*)&Vt[ti * 16 + nn][0];
#pragma unroll
            for (int jj = 0; jj < 8; ++jj) vf.d[jj] = vp[(kb2 >> 1) + jj];
            o[ti].v = WMMA_F16(pa.v, vf.v, o[ti].v);
        }
    }

    // normalize and store ctx as [B, S, H] f16
    const int rbase = (lane >> 4) * 8;
#pragma unroll
    for (int ti = 0; ti < 4; ++ti) {
#pragma unroll
        for (int r = 0; r < 8; ++r) {
            const int row = qt * 16 + rbase + r;
            const int col = hh * DH + ti * 16 + nn;
            ctx[((size_t)bi * Ss + row) * Hh + col] = (f16)(o[ti].f[r] / lrow[r]);
        }
    }
}

// ---------------------------------------------------------------------------
// host
// ---------------------------------------------------------------------------
extern "C" void kernel_launch(void* const* d_in, const int* in_sizes, int n_in,
                              void* d_out, int out_size, void* d_ws, size_t ws_size,
                              hipStream_t stream) {
    const float* h_in  = (const float*)d_in[0];
    const float* amask = (const float*)d_in[1];
    const float* Wq = (const float*)d_in[2];
    const float* bq = (const float*)d_in[3];
    const float* Wk = (const float*)d_in[4];
    const float* bk = (const float*)d_in[5];
    const float* Wv = (const float*)d_in[6];
    const float* bv = (const float*)d_in[7];
    const float* Wp = (const float*)d_in[8];
    const float* bp = (const float*)d_in[9];
    const float* g1 = (const float*)d_in[10];
    const float* be1 = (const float*)d_in[11];
    const float* W1 = (const float*)d_in[12];
    const float* bf1 = (const float*)d_in[13];
    const float* W2 = (const float*)d_in[14];
    const float* bf2 = (const float*)d_in[15];

    const size_t HH = (size_t)Hh * Hh;       // 589824
    const size_t HD = (size_t)Hh * DFF;      // 2359296
    const size_t ACT = (size_t)MROW * Hh;    // 6291456

    size_t off = 0;
    auto carve = [&](size_t bytes) {
        void* p = (char*)d_ws + off;
        off += (bytes + 255) & ~(size_t)255;
        return p;
    };
    float* hbuf  = (float*)carve(ACT * 4);               // hidden state f32
    f16*   xln   = (f16*)carve(ACT * 2);                 // LN output f16
    f16*   region = (f16*)carve(4 * ACT * 2);            // q,k,v,ctx  (aliased by ffn1)
    f16* qbuf = region;
    f16* kbuf = region + ACT;
    f16* vbuf = region + 2 * ACT;
    f16* cbuf = region + 3 * ACT;
    f16* ffn1 = region;                                   // 8192*3072 f16 == 4*ACT exactly
    f16* wq16 = (f16*)carve(HH * 2);
    f16* wk16 = (f16*)carve(HH * 2);
    f16* wv16 = (f16*)carve(HH * 2);
    f16* wp16 = (f16*)carve(HH * 2);
    f16* w116 = (f16*)carve(HD * 2);
    f16* w216 = (f16*)carve(HD * 2);

    hipMemcpyAsync(hbuf, h_in, ACT * 4, hipMemcpyDeviceToDevice, stream);

    for (int l = 0; l < 12; ++l) {
        cvt_kernel<<<HH / 256, 256, 0, stream>>>(Wq + l * HH, wq16, (int)HH);
        cvt_kernel<<<HH / 256, 256, 0, stream>>>(Wk + l * HH, wk16, (int)HH);
        cvt_kernel<<<HH / 256, 256, 0, stream>>>(Wv + l * HH, wv16, (int)HH);
        cvt_kernel<<<HH / 256, 256, 0, stream>>>(Wp + l * HH, wp16, (int)HH);
        cvt_kernel<<<HD / 256, 256, 0, stream>>>(W1 + l * HD, w116, (int)HD);
        cvt_kernel<<<HD / 256, 256, 0, stream>>>(W2 + l * HD, w216, (int)HD);

        ln_kernel<<<MROW, 256, 0, stream>>>(hbuf, g1 + l * Hh, be1 + l * Hh, xln);

        gemm_kernel<<<dim3(MROW / 256, Hh / 64), 256, 0, stream>>>(
            xln, wq16, bq + l * Hh, Hh, Hh, 1, qbuf, nullptr, nullptr);
        gemm_kernel<<<dim3(MROW / 256, Hh / 64), 256, 0, stream>>>(
            xln, wk16, bk + l * Hh, Hh, Hh, 1, kbuf, nullptr, nullptr);
        gemm_kernel<<<dim3(MROW / 256, Hh / 64), 256, 0, stream>>>(
            xln, wv16, bv + l * Hh, Hh, Hh, 1, vbuf, nullptr, nullptr);

        attn_kernel<<<dim3(Ss / 16, NH, Bb), 32, 0, stream>>>(qbuf, kbuf, vbuf, amask, cbuf);

        gemm_kernel<<<dim3(MROW / 256, Hh / 64), 256, 0, stream>>>(
            cbuf, wp16, bp + l * Hh, Hh, Hh, 2, nullptr, hbuf, hbuf);

        ln_kernel<<<MROW, 256, 0, stream>>>(hbuf, g1 + l * Hh, be1 + l * Hh, xln);

        gemm_kernel<<<dim3(MROW / 256, DFF / 64), 256, 0, stream>>>(
            xln, w116, bf1 + l * DFF, Hh, DFF, 3, ffn1, nullptr, nullptr);
        gemm_kernel<<<dim3(MROW / 256, Hh / 64), 256, 0, stream>>>(
            ffn1, w216, bf2 + l * Hh, DFF, Hh, 2, nullptr, hbuf, hbuf);
    }

    hipMemcpyAsync(d_out, hbuf, ACT * 4, hipMemcpyDeviceToDevice, stream);
}
